// Paraphraser_31533649887353
// MI455X (gfx1250) — compile-verified, matrix operand order checked by reading, not run on hardware
//
#include <hip/hip_runtime.h>
#include <hip/hip_bf16.h>

// ---------------------------------------------------------------------------
// CDNA5 (gfx1250) wave32 WMMA implementation of seq2seq LSTM + attention NMT
// ---------------------------------------------------------------------------

typedef __bf16 bf16;
typedef __attribute__((ext_vector_type(16))) __bf16 v16bf_t;
typedef __attribute__((ext_vector_type(8)))  __bf16 v8bf_t;
typedef __attribute__((ext_vector_type(8)))  float  v8f_t;

#define S_    40
#define B_    32
#define T_    40
#define E_    256
#define H_    512
#define VT_   50000
#define VTPAD 50176    // padded vocab rows so edge-tile loads stay in-bounds
#define G4H   2048     // 4*H
#define NROW  1248     // (T-1)*B
#define VNCHK 391      // ceil(50000/128)

__device__ __forceinline__ bf16 f2bf(float x) {
  unsigned u = __float_as_uint(x);
  u = (u + 0x7FFFu + ((u >> 16) & 1u)) >> 16;
  unsigned short s = (unsigned short)u;
  return __builtin_bit_cast(bf16, s);
}
__device__ __forceinline__ float sigmf(float x) { return 1.0f / (1.0f + expf(-x)); }

// gfx1250 async global->LDS copy (16B per lane), tracked by ASYNCcnt.
// The LDS offset is derived by ptrtoint of the real __shared__ pointer INSIDE
// this helper: that captures/escapes the LDS object, so together with the
// "memory" clobber LLVM must treat this asm as a potential writer of the
// staged buffer (otherwise the later ds_loads get folded to undef).
__device__ __forceinline__ void async_g2l_b128(void* lds, const void* gptr) {
  unsigned lds_off = (unsigned)(size_t)lds;   // flat LDS aperture: addr[31:0]
  asm volatile("global_load_async_to_lds_b128 %0, %1, off"
               :: "v"(lds_off), "v"(gptr)
               : "memory");
}
__device__ __forceinline__ void wait_async0() {
  asm volatile("s_wait_asynccnt 0x0" ::: "memory");
}
__device__ __forceinline__ void wait_async16() {   // allow 16 in flight
  asm volatile("s_wait_asynccnt 0x10" ::: "memory");
}

// Load a 16x32 (MxK) bf16 WMMA operand fragment for this lane.
// halves 0..7  : K = k0 + hi*8 .. +7
// halves 8..15 : K = k0 + 16 + hi*8 .. +7          (ISA 7.12.2, 16-bit A layout)
__device__ __forceinline__ v16bf_t load_frag(const bf16* rowp, int k0, int hi) {
  const bf16* p = rowp + k0 + hi * 8;
  v8bf_t lo = *(const v8bf_t*)(p);
  v8bf_t hh = *(const v8bf_t*)(p + 16);
  v16bf_t r;
#pragma unroll
  for (int i = 0; i < 8; ++i) { r[i] = lo[i]; r[i + 8] = hh[i]; }
  return r;
}
__device__ __forceinline__ v8f_t wmma_bf16(v16bf_t a, v16bf_t b, v8f_t c) {
  return __builtin_amdgcn_wmma_f32_16x16x32_bf16(false, a, false, b, (short)0, c,
                                                 false, false);
}

// Stage one K-chunk (32 rows x klen<=512 cols) of A into LDS (row stride 512).
// 16B per async transfer; per-thread transfer count = klen/32 (16 when klen=512).
__device__ __forceinline__ void stage_chunk(bf16* dst, const bf16* A, int lda,
                                            int m0, int kc, int K) {
  int klen = K - kc; if (klen > 512) klen = 512;
  int tpr = klen >> 3;                       // 16B transfers per row (pow2)
  int tlog = (tpr == 64) ? 6 : 5;
  for (int j = threadIdx.x; j < 32 * tpr; j += 128) {
    int row = j >> tlog, cb = j & (tpr - 1);
    async_g2l_b128(dst + row * 512 + cb * 8,
                   A + (size_t)(m0 + row) * lda + kc + cb * 8);
  }
}

// ---------------------------------------------------------------------------
// Generic GEMM:  C[M,N] = A[M,K](bf16) * Bm[N,K]^T(bf16)  (+ initC) (+ bias)
// Block = 4 waves (128 thr).  Block tile 32x128; per-wave 2x2 WMMA tiles.
// A tile (32 rows, shared by all 4 waves) is async-staged into LDS with
// double buffering: chunk c+1's async copies overlap chunk c's WMMAs.
// Requires M % 32 == 0 (true for all call sites).
// ---------------------------------------------------------------------------
__device__ __forceinline__ void store_tile(float* C, int ldc, const float* initC,
                                           const float* bias, int m0, int n0,
                                           int lane16, int hi, v8f_t acc) {
#pragma unroll
  for (int v = 0; v < 8; ++v) {
    int m = m0 + v + 8 * hi;
    int n = n0 + lane16;
    float val = acc[v];
    if (initC) val += initC[(size_t)m * ldc + n];
    if (bias)  val += bias[n];
    C[(size_t)m * ldc + n] = val;
  }
}

__global__ __launch_bounds__(128) void k_gemm(
    const bf16* A, int lda, const bf16* Bm, int ldb,
    float* C, int ldc, const float* initC, const float* bias,
    int M, int N, int K)
{
  __shared__ __align__(16) bf16 tA[2][32 * 512];   // double-buffered A chunks
  int wave = threadIdx.x >> 5, lane = threadIdx.x & 31;
  int lane16 = lane & 15, hi = lane >> 4;
  int n0 = blockIdx.x * 128 + wave * 32;
  int m0 = blockIdx.y * 32;
  if (m0 + 32 > M) return;                         // block-uniform exit only
  bool active = (n0 + 32 <= N);

  const bf16* brow0 = Bm + (size_t)(n0 + lane16) * ldb;
  const bf16* brow1 = brow0 + (size_t)16 * ldb;
  int nchunk = (K + 511) >> 9;

  stage_chunk(tA[0], A, lda, m0, 0, K);            // prologue: chunk 0
  v8f_t a00 = {}, a01 = {}, a10 = {}, a11 = {};
  for (int c = 0; c < nchunk; ++c) {
    int kc = c << 9;
    int klen = K - kc; if (klen > 512) klen = 512;
    bool has_next = (c + 1) < nchunk;
    if (has_next) {
      stage_chunk(tA[(c + 1) & 1], A, lda, m0, kc + 512, K);
      wait_async16();                              // retire chunk c only
    } else {
      wait_async0();
    }
    __syncthreads();                               // chunk c visible to all
    if (active) {
      const bf16* arow0 = tA[c & 1] + lane16 * 512;
      const bf16* arow1 = tA[c & 1] + (16 + lane16) * 512;
      for (int k0 = 0; k0 < klen; k0 += 32) {
        v16bf_t af0 = load_frag(arow0, k0, hi);
        v16bf_t af1 = load_frag(arow1, k0, hi);
        v16bf_t bf0 = load_frag(brow0, kc + k0, hi);
        v16bf_t bf1 = load_frag(brow1, kc + k0, hi);
        a00 = wmma_bf16(af0, bf0, a00);
        a01 = wmma_bf16(af0, bf1, a01);
        a10 = wmma_bf16(af1, bf0, a10);
        a11 = wmma_bf16(af1, bf1, a11);
      }
    }
    __syncthreads();              // reads done before next iter overwrites buf
  }
  if (active) {
    store_tile(C, ldc, initC, bias, m0,      n0,      lane16, hi, a00);
    store_tile(C, ldc, initC, bias, m0,      n0 + 16, lane16, hi, a01);
    store_tile(C, ldc, initC, bias, m0 + 16, n0,      lane16, hi, a10);
    store_tile(C, ldc, initC, bias, m0 + 16, n0 + 16, lane16, hi, a11);
  }
}

// ---------------------------------------------------------------------------
// Vocab projection with fused partial log-sum-exp (pass 1).
// Block: 32 rows x 128 vocab cols of logits (2x2 WMMA tiles per wave), A tile
// async-staged to LDS, then per-row (max, sum-exp) partials + target capture.
// W must be padded to VTPAD rows (values past VT_ are discarded via -inf).
// ---------------------------------------------------------------------------
__global__ __launch_bounds__(128) void k_vocab_partial(
    const bf16* A /*NROW x H*/, const bf16* W /*VTPAD x H*/,
    const int* tgt /*NROW*/, float* pmax, float* psum, float* plog)
{
  __shared__ __align__(16) char smem[32 * 512 * 2 + 32 * 129 * 4];
  bf16*  tA   = (bf16*)smem;                     // 32 x 512 (32 KB)
  float* tOut = (float*)(smem + 32 * 512 * 2);   // 32 x 128, stride 129

  int wave = threadIdx.x >> 5, lane = threadIdx.x & 31;
  int lane16 = lane & 15, hi = lane >> 4;
  int m0    = blockIdx.y * 32;
  int chunk = blockIdx.x;
  int nbase = chunk * 128;
  int n0w   = nbase + wave * 32;
  bool val0 = (n0w + 16) <= VT_;
  bool val1 = (n0w + 32) <= VT_;

  // async-stage A tile: rows m0..m0+31 are contiguous 32 KB (lda == H_)
  {
    const char* gsrc = (const char*)(A + (size_t)m0 * H_);
    char* ldst = (char*)tA;
    for (int it = 0; it < 16; ++it) {
      unsigned off = (unsigned)(threadIdx.x + it * 128) * 16u;
      async_g2l_b128(ldst + off, gsrc + off);
    }
    wait_async0();
    __syncthreads();
  }

  const bf16* arow0 = tA + lane16 * 512;
  const bf16* arow1 = tA + (16 + lane16) * 512;
  const bf16* brow0 = W + (size_t)(n0w + lane16) * H_;
  const bf16* brow1 = brow0 + (size_t)16 * H_;
  v8f_t a00 = {}, a01 = {}, a10 = {}, a11 = {};
  for (int k0 = 0; k0 < H_; k0 += 32) {
    v16bf_t af0 = load_frag(arow0, k0, hi);
    v16bf_t af1 = load_frag(arow1, k0, hi);
    v16bf_t bf0 = load_frag(brow0, k0, hi);
    v16bf_t bf1 = load_frag(brow1, k0, hi);
    a00 = wmma_bf16(af0, bf0, a00);
    a01 = wmma_bf16(af0, bf1, a01);
    a10 = wmma_bf16(af1, bf0, a10);
    a11 = wmma_bf16(af1, bf1, a11);
  }
  const float ninf = -__builtin_inff();
#pragma unroll
  for (int v = 0; v < 8; ++v) {
    int c0 = wave * 32 + lane16, c1 = c0 + 16;
    tOut[(v + 8 * hi) * 129 + c0]      = val0 ? a00[v] : ninf;
    tOut[(v + 8 * hi) * 129 + c1]      = val1 ? a01[v] : ninf;
    tOut[(16 + v + 8 * hi) * 129 + c0] = val0 ? a10[v] : ninf;
    tOut[(16 + v + 8 * hi) * 129 + c1] = val1 ? a11[v] : ninf;
  }
  __syncthreads();
  if (threadIdx.x < 32) {
    int r = m0 + threadIdx.x;
    const float* rowp = tOut + threadIdx.x * 129;
    float mx = ninf;
    for (int j = 0; j < 128; ++j) mx = fmaxf(mx, rowp[j]);
    float s = 0.f;
    for (int j = 0; j < 128; ++j) s += expf(rowp[j] - mx);
    pmax[(size_t)r * VNCHK + chunk] = mx;
    psum[(size_t)r * VNCHK + chunk] = s;
    int tg = tgt[r];
    if (tg >= nbase && tg < nbase + 128) plog[r] = rowp[tg - nbase];
  }
}

// Pass 2: merge chunk partials -> picked[r] = logit[tgt] - logsumexp  (masked)
__global__ __launch_bounds__(256) void k_vocab_reduce(
    const float* pmax, const float* psum, const float* plog,
    const int* tgt, float* picked)
{
  int r = blockIdx.x;
  __shared__ float sm[256];
  float mx = -__builtin_inff();
  for (int c = threadIdx.x; c < VNCHK; c += 256)
    mx = fmaxf(mx, pmax[(size_t)r * VNCHK + c]);
  sm[threadIdx.x] = mx; __syncthreads();
  for (int st = 128; st > 0; st >>= 1) {
    if (threadIdx.x < st) sm[threadIdx.x] = fmaxf(sm[threadIdx.x], sm[threadIdx.x + st]);
    __syncthreads();
  }
  float gmax = sm[0]; __syncthreads();
  float s = 0.f;
  for (int c = threadIdx.x; c < VNCHK; c += 256)
    s += expf(pmax[(size_t)r * VNCHK + c] - gmax) * psum[(size_t)r * VNCHK + c];
  sm[threadIdx.x] = s; __syncthreads();
  for (int st = 128; st > 0; st >>= 1) {
    if (threadIdx.x < st) sm[threadIdx.x] += sm[threadIdx.x + st];
    __syncthreads();
  }
  if (threadIdx.x == 0) {
    float logZ = gmax + logf(sm[0]);
    picked[r] = (tgt[r] != 0 /*PAD*/) ? (plog[r] - logZ) : 0.f;
  }
}

__global__ void k_final(const float* picked, float* out) {
  int b = threadIdx.x;
  if (b < B_) {
    float s = 0.f;
    for (int t = 0; t < T_ - 1; ++t) s += picked[t * B_ + b];
    out[b] = s;
  }
}

// ---------------------------------------------------------------------------
// Elementwise / small kernels
// ---------------------------------------------------------------------------
__global__ void k_cvt(const float* s, bf16* d, size_t n) {
  size_t i = (size_t)blockIdx.x * blockDim.x + threadIdx.x;
  if (i < n) d[i] = f2bf(s[i]);
}
__global__ void k_zero_bf16(bf16* p, int n) {
  int i = blockIdx.x * blockDim.x + threadIdx.x;
  if (i < n) p[i] = f2bf(0.f);
}
__global__ void k_zero_f32(float* p, int n) {
  int i = blockIdx.x * blockDim.x + threadIdx.x;
  if (i < n) p[i] = 0.f;
}
__global__ void k_embed(const float* tbl, const int* ids, bf16* outp, int rows) {
  int idx = blockIdx.x * blockDim.x + threadIdx.x;
  if (idx >= rows * E_) return;
  int r = idx / E_, e = idx - r * E_;
  outp[idx] = f2bf(tbl[(size_t)ids[r] * E_ + e]);
}
// LSTM cell: gates (B x 4H, order i,f,g,o) ; c updated in place; h -> f32+bf16
__global__ void k_lstm_cell(const float* g, float* c, float* h_out, bf16* h_bf) {
  int i = blockIdx.x * blockDim.x + threadIdx.x;
  if (i >= B_ * H_) return;
  int b = i >> 9, j = i & (H_ - 1);
  const float* gb = g + (size_t)b * G4H;
  float gi = gb[j], gf = gb[H_ + j], gg = gb[2 * H_ + j], go = gb[3 * H_ + j];
  float cv = sigmf(gf) * c[i] + sigmf(gi) * tanhf(gg);
  c[i] = cv;
  float hv = sigmf(go) * tanhf(cv);
  h_out[i] = hv;
  h_bf[i] = f2bf(hv);
}
// enc_hidden[b][s][0:H]=hs_f[s][b]; [H:2H]=hs_b[s][b]  (f32 + bf16)
__global__ void k_build_ench(const float* hs_f, const float* hs_b,
                             float* ench, bf16* ench_bf) {
  int idx = blockIdx.x * blockDim.x + threadIdx.x;
  if (idx >= B_ * S_ * 2 * H_) return;
  int d = idx & (2 * H_ - 1);
  int bs = idx >> 10;
  int s = bs % S_, b = bs / S_;
  float v = (d < H_) ? hs_f[((size_t)s * B_ + b) * H_ + d]
                     : hs_b[((size_t)s * B_ + b) * H_ + (d - H_)];
  ench[idx] = v;
  ench_bf[idx] = f2bf(v);
}
__global__ void k_concat2_bf16(const float* x0, int n0, const float* x1, int n1,
                               bf16* outp, int rows) {
  int idx = blockIdx.x * blockDim.x + threadIdx.x;
  int w = n0 + n1;
  if (idx >= rows * w) return;
  int r = idx / w, cc = idx - r * w;
  float v = (cc < n0) ? x0[(size_t)r * n0 + cc] : x1[(size_t)r * n1 + (cc - n0)];
  outp[idx] = f2bf(v);
}
// scores[b][s] = dot(enc_proj[b*S+s], h[b])  (masked)
__global__ __launch_bounds__(128) void k_scores(const float* encp, const float* h,
                                                const float* masks, float* scores) {
  int bs = blockIdx.x;
  int b = bs / S_, s = bs - b * S_;
  __shared__ float red[128];
  const float* ep = encp + (size_t)bs * H_;
  const float* hp = h + (size_t)b * H_;
  float acc = 0.f;
  for (int j = threadIdx.x; j < H_; j += 128) acc += ep[j] * hp[j];
  red[threadIdx.x] = acc; __syncthreads();
  for (int st = 64; st > 0; st >>= 1) {
    if (threadIdx.x < st) red[threadIdx.x] += red[threadIdx.x + st];
    __syncthreads();
  }
  if (threadIdx.x == 0)
    scores[bs] = (masks[b * S_ + s] > 0.f) ? -__builtin_inff() : red[0];
}
__global__ __launch_bounds__(64) void k_softmax40(const float* scores, float* att) {
  int b = blockIdx.x, t = threadIdx.x;
  __shared__ float sh[64];
  float v = (t < S_) ? scores[b * S_ + t] : -__builtin_inff();
  sh[t] = v; __syncthreads();
  for (int st = 32; st > 0; st >>= 1) {
    if (t < st) sh[t] = fmaxf(sh[t], sh[t + st]);
    __syncthreads();
  }
  float mx = sh[0]; __syncthreads();
  float e = (t < S_) ? expf(v - mx) : 0.f;
  sh[t] = e; __syncthreads();
  for (int st = 32; st > 0; st >>= 1) {
    if (t < st) sh[t] += sh[t + st];
    __syncthreads();
  }
  if (t < S_) att[b * S_ + t] = e / sh[0];
}
__global__ void k_ctx(const float* att, const float* ench, float* ctx) {
  int idx = blockIdx.x * blockDim.x + threadIdx.x;  // b*1024 + d
  if (idx >= B_ * 2 * H_) return;
  int b = idx >> 10, d = idx & (2 * H_ - 1);
  float acc = 0.f;
  for (int s = 0; s < S_; ++s)
    acc += att[b * S_ + s] * ench[((size_t)b * S_ + s) * (2 * H_) + d];
  ctx[idx] = acc;
}
__global__ void k_tanh_o(const float* pre, bf16* o_bf, bf16* outs_slot) {
  int i = blockIdx.x * blockDim.x + threadIdx.x;
  if (i >= B_ * H_) return;
  bf16 b = f2bf(tanhf(pre[i]));
  o_bf[i] = b;
  outs_slot[i] = b;
}

// ---------------------------------------------------------------------------
// Host orchestration
// ---------------------------------------------------------------------------
extern "C" void kernel_launch(void* const* d_in, const int* in_sizes, int n_in,
                              void* d_out, int out_size, void* d_ws, size_t ws_size,
                              hipStream_t stream)
{
  const int*   src_ids = (const int*)  d_in[0];
  const int*   tgt_ids = (const int*)  d_in[1];
  const float* masks   = (const float*)d_in[2];
  const float* src_emb = (const float*)d_in[3];
  const float* tgt_emb = (const float*)d_in[4];
  const float* eWih_f  = (const float*)d_in[5];
  const float* eWhh_f  = (const float*)d_in[6];
  const float* eb_f    = (const float*)d_in[7];
  const float* eWih_b  = (const float*)d_in[8];
  const float* eWhh_b  = (const float*)d_in[9];
  const float* eb_b    = (const float*)d_in[10];
  const float* dWih    = (const float*)d_in[11];
  const float* dWhh    = (const float*)d_in[12];
  const float* db      = (const float*)d_in[13];
  const float* W_hp    = (const float*)d_in[14];
  const float* W_cp    = (const float*)d_in[15];
  const float* W_attp  = (const float*)d_in[16];
  const float* W_combp = (const float*)d_in[17];
  const float* W_vocp  = (const float*)d_in[18];
  float* out = (float*)d_out;

  // ---- workspace carve ----
  char* wsb = (char*)d_ws;
  size_t off = 0;
  auto alloc = [&](size_t bytes) -> void* {
    void* p = wsb + off;
    off = (off + bytes + 255) & ~(size_t)255;
    return p;
  };
  bf16* wihf_b  = (bf16*)alloc((size_t)G4H * E_ * 2);
  bf16* whhf_b  = (bf16*)alloc((size_t)G4H * H_ * 2);
  bf16* wihb_b  = (bf16*)alloc((size_t)G4H * E_ * 2);
  bf16* whhb_b  = (bf16*)alloc((size_t)G4H * H_ * 2);
  bf16* dwih_b  = (bf16*)alloc((size_t)G4H * (E_ + H_) * 2);
  bf16* dwhh_b  = (bf16*)alloc((size_t)G4H * H_ * 2);
  bf16* wh_b    = (bf16*)alloc((size_t)H_ * 2 * H_ * 2);
  bf16* wc_b    = (bf16*)alloc((size_t)H_ * 2 * H_ * 2);
  bf16* watt_b  = (bf16*)alloc((size_t)H_ * 2 * H_ * 2);
  bf16* wcomb_b = (bf16*)alloc((size_t)H_ * 3 * H_ * 2);
  bf16* wvoc_b  = (bf16*)alloc((size_t)VTPAD * H_ * 2);   // padded rows
  bf16* xsrc    = (bf16*)alloc((size_t)S_ * B_ * E_ * 2);
  bf16* xtgt    = (bf16*)alloc((size_t)NROW * E_ * 2);
  bf16* h_bf    = (bf16*)alloc((size_t)B_ * H_ * 2);
  bf16* o_bf    = (bf16*)alloc((size_t)B_ * H_ * 2);
  bf16* xc_bf   = (bf16*)alloc((size_t)B_ * 3 * H_ * 2);
  bf16* cat_bf  = (bf16*)alloc((size_t)B_ * 2 * H_ * 2);
  bf16* ench_bf = (bf16*)alloc((size_t)B_ * S_ * 2 * H_ * 2);
  bf16* outs_bf = (bf16*)alloc((size_t)NROW * H_ * 2);

  float* XW_f  = (float*)alloc((size_t)S_ * B_ * G4H * 4);
  float* XW_b  = (float*)alloc((size_t)S_ * B_ * G4H * 4);
  float* XW_d  = (float*)alloc((size_t)NROW * G4H * 4);
  float* gates = (float*)alloc((size_t)B_ * G4H * 4);
  float* hs_f  = (float*)alloc((size_t)S_ * B_ * H_ * 4);
  float* hs_b  = (float*)alloc((size_t)S_ * B_ * H_ * 4);
  float* c_f   = (float*)alloc((size_t)B_ * H_ * 4);
  float* c_b   = (float*)alloc((size_t)B_ * H_ * 4);
  float* h_dec = (float*)alloc((size_t)B_ * H_ * 4);
  float* c_dec = (float*)alloc((size_t)B_ * H_ * 4);
  float* ench  = (float*)alloc((size_t)B_ * S_ * 2 * H_ * 4);
  float* encp  = (float*)alloc((size_t)B_ * S_ * H_ * 4);
  float* ctx   = (float*)alloc((size_t)B_ * 2 * H_ * 4);
  float* scor  = (float*)alloc((size_t)B_ * S_ * 4);
  float* att   = (float*)alloc((size_t)B_ * S_ * 4);
  float* pre_o = (float*)alloc((size_t)B_ * H_ * 4);
  float* pmax  = (float*)alloc((size_t)NROW * VNCHK * 4);
  float* psum  = (float*)alloc((size_t)NROW * VNCHK * 4);
  float* plog  = (float*)alloc((size_t)NROW * 4);
  float* pick  = (float*)alloc((size_t)NROW * 4);

  auto cvt = [&](const float* s, bf16* d, size_t n) {
    k_cvt<<<dim3((unsigned)((n + 255) / 256)), dim3(256), 0, stream>>>(s, d, n);
  };
  auto gemm = [&](const bf16* A, int lda, const bf16* Bm, int ldb,
                  float* C, int ldc, const float* initC, const float* bias,
                  int M, int N, int K) {
    dim3 g((N + 127) / 128, (M + 31) / 32);
    k_gemm<<<g, 128, 0, stream>>>(A, lda, Bm, ldb, C, ldc, initC, bias, M, N, K);
  };

  // ---- weight conversion (f32 -> bf16) ----
  cvt(eWih_f, wihf_b, (size_t)G4H * E_);
  cvt(eWhh_f, whhf_b, (size_t)G4H * H_);
  cvt(eWih_b, wihb_b, (size_t)G4H * E_);
  cvt(eWhh_b, whhb_b, (size_t)G4H * H_);
  cvt(dWih,   dwih_b, (size_t)G4H * (E_ + H_));
  cvt(dWhh,   dwhh_b, (size_t)G4H * H_);
  cvt(W_hp,   wh_b,   (size_t)H_ * 2 * H_);
  cvt(W_cp,   wc_b,   (size_t)H_ * 2 * H_);
  cvt(W_attp, watt_b, (size_t)H_ * 2 * H_);
  cvt(W_combp, wcomb_b, (size_t)H_ * 3 * H_);
  cvt(W_vocp, wvoc_b, (size_t)VT_ * H_);

  // ---- embeddings ----
  k_embed<<<dim3((S_ * B_ * E_) / 256), 256, 0, stream>>>(src_emb, src_ids, xsrc, S_ * B_);
  k_embed<<<dim3((NROW * E_) / 256), 256, 0, stream>>>(tgt_emb, tgt_ids, xtgt, NROW);

  // ---- encoder input-side GEMMs (batched over all timesteps, bias fused) ----
  gemm(xsrc, E_, wihf_b, E_, XW_f, G4H, nullptr, eb_f, S_ * B_, G4H, E_);
  gemm(xsrc, E_, wihb_b, E_, XW_b, G4H, nullptr, eb_b, S_ * B_, G4H, E_);
  // decoder input-side GEMM: tgt_e @ dec_Wih[:, :E]^T + b  (ldb = E+H)
  gemm(xtgt, E_, dwih_b, E_ + H_, XW_d, G4H, nullptr, db, NROW, G4H, E_);

  // ---- forward encoder ----
  k_zero_bf16<<<64, 256, 0, stream>>>(h_bf, B_ * H_);
  k_zero_f32<<<64, 256, 0, stream>>>(c_f, B_ * H_);
  for (int s = 0; s < S_; ++s) {
    gemm(h_bf, H_, whhf_b, H_, gates, G4H, XW_f + (size_t)s * B_ * G4H, nullptr,
         B_, G4H, H_);
    k_lstm_cell<<<64, 256, 0, stream>>>(gates, c_f, hs_f + (size_t)s * B_ * H_, h_bf);
  }
  // ---- backward encoder (scan over reversed sequence; store at original s) ----
  k_zero_bf16<<<64, 256, 0, stream>>>(h_bf, B_ * H_);
  k_zero_f32<<<64, 256, 0, stream>>>(c_b, B_ * H_);
  for (int k = 0; k < S_; ++k) {
    int s = S_ - 1 - k;
    gemm(h_bf, H_, whhb_b, H_, gates, G4H, XW_b + (size_t)s * B_ * G4H, nullptr,
         B_, G4H, H_);
    k_lstm_cell<<<64, 256, 0, stream>>>(gates, c_b, hs_b + (size_t)s * B_ * H_, h_bf);
  }

  // ---- enc_hidden (B,S,2H) + bf16 copy; enc_proj = enc_hidden @ W_att^T ----
  k_build_ench<<<(B_ * S_ * 2 * H_) / 256, 256, 0, stream>>>(hs_f, hs_b, ench, ench_bf);
  gemm(ench_bf, 2 * H_, watt_b, 2 * H_, encp, H_, nullptr, nullptr,
       B_ * S_, H_, 2 * H_);

  // ---- decoder init: dec_h = [hT_f,hT_b]@W_h^T ; dec_c = [cT_f,cT_b]@W_c^T ----
  k_concat2_bf16<<<(B_ * 2 * H_ + 255) / 256, 256, 0, stream>>>(
      hs_f + (size_t)(S_ - 1) * B_ * H_, H_, hs_b /* s=0 = final bwd state */, H_,
      cat_bf, B_);
  gemm(cat_bf, 2 * H_, wh_b, 2 * H_, h_dec, H_, nullptr, nullptr, B_, H_, 2 * H_);
  k_concat2_bf16<<<(B_ * 2 * H_ + 255) / 256, 256, 0, stream>>>(c_f, H_, c_b, H_,
                                                                cat_bf, B_);
  gemm(cat_bf, 2 * H_, wc_b, 2 * H_, c_dec, H_, nullptr, nullptr, B_, H_, 2 * H_);
  cvt(h_dec, h_bf, (size_t)B_ * H_);
  k_zero_bf16<<<64, 256, 0, stream>>>(o_bf, B_ * H_);

  // ---- decoder steps ----
  for (int t = 0; t < T_ - 1; ++t) {
    // gates = XW_d[t] + o_prev @ dec_Wih[:,E:]^T + h_prev @ dec_Whh^T
    gemm(o_bf, H_, dwih_b + E_, E_ + H_, gates, G4H,
         XW_d + (size_t)t * B_ * G4H, nullptr, B_, G4H, H_);
    gemm(h_bf, H_, dwhh_b, H_, gates, G4H, gates, nullptr, B_, G4H, H_);
    k_lstm_cell<<<64, 256, 0, stream>>>(gates, c_dec, h_dec, h_bf);
    // attention
    k_scores<<<B_ * S_, 128, 0, stream>>>(encp, h_dec, masks, scor);
    k_softmax40<<<B_, 64, 0, stream>>>(scor, att);
    k_ctx<<<(B_ * 2 * H_) / 256, 256, 0, stream>>>(att, ench, ctx);
    // o_t = tanh([h, ctx] @ W_comb^T)
    k_concat2_bf16<<<(B_ * 3 * H_ + 255) / 256, 256, 0, stream>>>(h_dec, H_, ctx,
                                                                  2 * H_, xc_bf, B_);
    gemm(xc_bf, 3 * H_, wcomb_b, 3 * H_, pre_o, H_, nullptr, nullptr, B_, H_, 3 * H_);
    k_tanh_o<<<64, 256, 0, stream>>>(pre_o, o_bf, outs_bf + (size_t)t * B_ * H_);
  }

  // ---- vocab projection + fused log-softmax gather ----
  // tgt for row r=(t*B+b) is padded_target[(t+1)*B+b] = tgt_ids[r + B]
  k_vocab_partial<<<dim3(VNCHK, NROW / 32), 128, 0, stream>>>(
      outs_bf, wvoc_b, tgt_ids + B_, pmax, psum, plog);
  k_vocab_reduce<<<NROW, 256, 0, stream>>>(pmax, psum, plog, tgt_ids + B_, pick);
  k_final<<<1, 32, 0, stream>>>(pick, out);
  (void)in_sizes; (void)n_in; (void)out_size; (void)ws_size;
}
